// Model_1159641170370
// MI455X (gfx1250) — matrix operand order repaired and checked by reading, not compiled, not run on hardware
//
#include <hip/hip_runtime.h>
#include <hip/hip_bf16.h>
#include <math.h>

// ---------------- model constants ----------------
#define BATCH   16
#define LSEQ    512
#define VVARS   32
#define NSEQ    512          // B*V
#define DMODEL  128
#define DINNER  256
#define DSTATE  16
#define DTRANK  8
#define DCONV   4
#define PRED    96
#define PREDP   128          // padded head N
#define XPN     40
#define XPNP    64           // padded xproj N
#define SUMP    224
#define HEADNF  28672        // DMODEL * SUMP
#define SEH     32
#define KCHUNK  128          // TDM A-panel chunk (all GEMM K are multiples)
#define AROWB   272          // LDS A row pitch bytes: 256 + 16 pad (TDM pad)

// ---------------- types / helpers ----------------
typedef __attribute__((ext_vector_type(16))) __bf16 v16bf;
typedef __attribute__((ext_vector_type(8)))  float  v8f;
typedef __attribute__((ext_vector_type(4)))  unsigned v4u;
typedef __attribute__((ext_vector_type(8)))  int      v8i;
typedef __attribute__((ext_vector_type(4)))  int      v4i;

__device__ __forceinline__ unsigned short f2bfu(float f) {
    unsigned u = __builtin_bit_cast(unsigned, f);
    unsigned r = u + 0x7FFFu + ((u >> 16) & 1u);   // round-to-nearest-even
    return (unsigned short)(r >> 16);
}
__device__ __forceinline__ float bf2f(unsigned short h) {
    unsigned u = ((unsigned)h) << 16;
    return __builtin_bit_cast(float, u);
}

// ---------------------------------------------------------------------
// TDM: DMA a 64 x KCHUNK bf16 tile (row pitch lda elements) from global
// into LDS at lds_byte_addr, with 16B padding every 256B so the LDS tile
// has a 272B row pitch (bank-staggered).  D# built per CDNA5 ISA §8.3/8.4.
// Tracked with TENSORcnt; caller waits s_wait_tensorcnt + barrier.
// ---------------------------------------------------------------------
__device__ __forceinline__ void tdm_load_a_tile(const unsigned short* gptr,
                                                unsigned lds_byte_addr, int lda)
{
    unsigned long long ga = (unsigned long long)(const void*)gptr;
    v4u g0;
    g0[0] = 1u;                                   // count=1, user descriptor
    g0[1] = lds_byte_addr;                        // lds_addr
    g0[2] = (unsigned)ga;                         // global_addr[31:0]
    g0[3] = (unsigned)(ga >> 32) | (2u << 30);    // global_addr[56:32] | type=2
    v8i g1;
    g1[0] = (int)((1u << 16)        // data_size = 2 bytes
                | (1u << 20)        // pad_enable
                | (5u << 22)        // pad_interval: 64 DWORDs (256B)
                | (3u << 25));      // pad_amount:   4 DWORDs (16B)
    g1[1] = (int)(((unsigned)KCHUNK & 0xFFFFu) << 16);   // tensor_dim0 lo (abar=0)
    g1[2] = (int)(((unsigned)KCHUNK >> 16) | (64u << 16)); // td0 hi | tensor_dim1 lo
    g1[3] = (int)(((unsigned)KCHUNK) << 16);             // td1 hi=0 | tile_dim0
    g1[4] = (int)64u;                                    // tile_dim1=64, tile_dim2=0
    g1[5] = (int)(unsigned)lda;                          // tensor_dim0_stride lo
    g1[6] = 0;                                           // stride hi | td1_stride lo
    g1[7] = 0;
    v4i gz = {0, 0, 0, 0};                               // groups 2/3 unused (2D)
#if __has_include(<hip/amd_detail/amd_gfx1250_TDM.h>)
    // therock / clang-23 toolchain: 6-arg builtin
    v8i gz8 = {0, 0, 0, 0, 0, 0, 0, 0};
    __builtin_amdgcn_tensor_load_to_lds(g0, g1, gz, gz, gz8, 0);
#else
    // ROCm 7.2 / clang-22 toolchain: 5-arg builtin
    __builtin_amdgcn_tensor_load_to_lds(g0, g1, gz, gz, 0);
#endif
}

// =====================================================================
// bf16 GEMM via WMMA, guard-free (all dims padded to tile multiples):
//   C(MxN) = A(MxK) @ B(KxN), fp32 accumulate.
// Requirements: M%64==0, N%64==0, K%KCHUNK==0, lda/ldb%8==0, 16B ptrs.
// Block: 256 threads = 8 waves, 64x64 block tile, BK=32.
//  - A panel staged by the TENSOR DATA MOVER in KCHUNK slabs, double-
//    buffered in LDS (272B row pitch), overlapped with compute; waves
//    read fragments as two ds_load_b128 per lane (contiguous k = 16h + s order, shared with B,
//    keeps each lane's half-K in 32 contiguous bytes).
//  - B tile double-buffered in LDS, stored transposed (NxK) so fragment
//    reads are contiguous ds_load_b32; one barrier per K-step.
//  - Each wave owns one 16-row strip and two 16-col strips: A-frag reuse,
//    2 x v_wmma_f32_16x16x32_bf16 per K-step.
// mode 0: Cf = acc   mode 1: Cf += acc   mode 2: Cb = bf16(acc)
// =====================================================================
__global__ __launch_bounds__(256) void gemm_bf16_kernel(
    const unsigned short* __restrict__ A, int lda,
    const unsigned short* __restrict__ Bw, int ldb,
    float* __restrict__ Cf, unsigned short* __restrict__ Cb, int ldc,
    int K, int mode)
{
    __shared__ unsigned short AsL[2][64 * (AROWB / 2)];  // 2 x 64 x 136
    __shared__ unsigned short BsT[2][64][40];            // [buf][N][K] transposed

    const int tid  = threadIdx.x;
    const int bm   = blockIdx.y * 64;
    const int bn   = blockIdx.x * 64;
    const int lane = tid & 31;
    const int wv   = tid >> 5;
    const int mt   = wv & 3;        // M tile 0..3
    const int nh   = wv >> 2;       // N half 0..1 (two 16-col tiles each)

    union { v8f v; float f[8]; } acc0, acc1;
#pragma unroll
    for (int i = 0; i < 8; i++) { acc0.f[i] = 0.f; acc1.f[i] = 0.f; }

    // B tile loader mapping: 32(K) x 64(N), 8 contiguous N per thread
    const int br  = tid >> 3;          // K row within tile
    const int bc0 = (tid & 7) * 8;     // N col start

    const unsigned short* Apanel = A + (size_t)bm * lda;
    const unsigned ldsA0 = (unsigned)(size_t)(const void*)&AsL[0][0];
    const unsigned ldsA1 = (unsigned)(size_t)(const void*)&AsL[1][0];

    // issue TDM for A chunk 0 (wave 0 only; TDM is per-wave, EXEC-ignored)
    if (wv == 0) tdm_load_a_tile(Apanel, ldsA0, lda);

    // preload first B tile into buffer 0
    {
        uint4 b0 = *(const uint4*)(Bw + (size_t)br * ldb + bn + bc0);
        const unsigned short* bs = (const unsigned short*)&b0;
#pragma unroll
        for (int j = 0; j < 8; j++) BsT[0][bc0 + j][br] = bs[j];
    }
    __builtin_amdgcn_s_wait_tensorcnt(0);   // no-op for non-issuing waves
    __syncthreads();

    int bufA = 0, bufB = 0;
    const int kb = (lane >> 4) * 16;          // K base for B fragment
    const int n0 = nh * 32 + (lane & 15);     // first N-tile column
    const int nchunk = K / KCHUNK;

    // per-lane A fragment base inside an LDS A slab (byte offset)
    const int aoff = (mt * 16 + (lane & 15)) * AROWB + ((lane >> 4) * 16) * 2;

    for (int c = 0; c < nchunk; ++c) {
        // DMA next A slab into the other buffer while computing this one
        if (c + 1 < nchunk && wv == 0)
            tdm_load_a_tile(Apanel + (size_t)(c + 1) * KCHUNK,
                            bufA ? ldsA0 : ldsA1, lda);

        const char* abase = (const char*)&AsL[bufA][0] + aoff;

        for (int kk = 0; kk < KCHUNK; kk += 32) {
            const int k0 = c * KCHUNK + kk;

            // ---- A fragment: two b128 LDS loads from the TDM-staged slab
            union { v16bf v; unsigned u[8]; } fa, fb0, fb1;
            {
                uint4 a0 = *(const uint4*)(abase + kk * 2);
                uint4 a1 = *(const uint4*)(abase + kk * 2 + 16);
                fa.u[0] = a0.x; fa.u[1] = a0.y; fa.u[2] = a0.z; fa.u[3] = a0.w;
                fa.u[4] = a1.x; fa.u[5] = a1.y; fa.u[6] = a1.z; fa.u[7] = a1.w;
            }
            // ---- B fragments from LDS (contiguous 32-bit reads)
#pragma unroll
            for (int j = 0; j < 8; j++) {
                fb0.u[j] = *(const unsigned*)&BsT[bufB][n0][kb + 2 * j];
                fb1.u[j] = *(const unsigned*)&BsT[bufB][n0 + 16][kb + 2 * j];
            }
            // ---- prefetch next B tile into the other buffer (uniform branch)
            if (k0 + 32 < K) {
                uint4 bnx = *(const uint4*)(Bw + (size_t)(k0 + 32 + br) * ldb + bn + bc0);
                const unsigned short* bs = (const unsigned short*)&bnx;
#pragma unroll
                for (int j = 0; j < 8; j++) BsT[bufB ^ 1][bc0 + j][br] = bs[j];
            }

            acc0.v = __builtin_amdgcn_wmma_f32_16x16x32_bf16(
                false, fa.v, false, fb0.v, (short)0, acc0.v, false, false);
            acc1.v = __builtin_amdgcn_wmma_f32_16x16x32_bf16(
                false, fa.v, false, fb1.v, (short)0, acc1.v, false, false);

            __syncthreads();
            bufB ^= 1;
        }

        // make next A slab visible before flipping buffers
        __builtin_amdgcn_s_wait_tensorcnt(0);
        __syncthreads();
        bufA ^= 1;
    }

    // C/D layout: lanes 0-15 -> N=lane, rows M 0..7; lanes 16-31 -> rows 8..15
    const int nbase = bn + nh * 32 + (lane & 15);
    const int mbase = bm + mt * 16 + ((lane >> 4) ? 8 : 0);
#pragma unroll
    for (int i = 0; i < 8; i++) {
        size_t o0 = (size_t)(mbase + i) * ldc + nbase;
        if (mode == 0) {
            Cf[o0] = acc0.f[i];           Cf[o0 + 16] = acc1.f[i];
        } else if (mode == 1) {
            Cf[o0] += acc0.f[i];          Cf[o0 + 16] += acc1.f[i];
        } else {
            Cb[o0] = f2bfu(acc0.f[i]);    Cb[o0 + 16] = f2bfu(acc1.f[i]);
        }
    }
}

// =====================================================================
// Per-(b,v) instance norm stats + normalized, transposed series xn[n][l]
// =====================================================================
__global__ __launch_bounds__(256) void normalize_kernel(
    const float* __restrict__ x_enc, float* __restrict__ xn,
    float* __restrict__ meanv, float* __restrict__ stdv)
{
    const int n = blockIdx.x, b = n >> 5, v = n & 31;
    const int tid = threadIdx.x;
    __shared__ float red[256];
    float s = 0.f, s2 = 0.f;
    for (int l = tid; l < LSEQ; l += 256) {
        float x = x_enc[((size_t)b * LSEQ + l) * VVARS + v];
        s += x; s2 += x * x;
    }
    red[tid] = s; __syncthreads();
    for (int o = 128; o; o >>= 1) { if (tid < o) red[tid] += red[tid + o]; __syncthreads(); }
    float mean = red[0] / (float)LSEQ; __syncthreads();
    red[tid] = s2; __syncthreads();
    for (int o = 128; o; o >>= 1) { if (tid < o) red[tid] += red[tid + o]; __syncthreads(); }
    float var = red[0] / (float)LSEQ - mean * mean;
    float sd  = sqrtf(var + 1e-5f);
    if (tid == 0) { meanv[n] = mean; stdv[n] = sd; }
    float inv = 1.f / sd;
    for (int l = tid; l < LSEQ; l += 256) {
        float x = x_enc[((size_t)b * LSEQ + l) * VVARS + v];
        xn[(size_t)n * LSEQ + l] = (x - mean) * inv;
    }
}

// f32 -> bf16 with optional column padding (cols cs..cd-1 zero-filled)
__global__ void convert_pad_bf16(const float* __restrict__ src,
                                 unsigned short* __restrict__ dst,
                                 int rows, int cs, int cd)
{
    int i = blockIdx.x * 256 + threadIdx.x;
    if (i < rows * cd) {
        int r = i / cd, c = i - r * cd;
        dst[i] = (c < cs) ? f2bfu(src[(size_t)r * cs + c]) : (unsigned short)0;
    }
}

// =====================================================================
// Patch embed: z[n][t][d] = sum_k xn_pad[t*s+k] * val_w[k][d] + PE(t,d)
// =====================================================================
__global__ __launch_bounds__(128) void patch_embed_kernel(
    const float* __restrict__ xn, const float* __restrict__ valw,
    float* __restrict__ z, int Pn, int plen, int pstr)
{
    const int bid = blockIdx.x;
    const int n = bid / Pn, t = bid % Pn, d = threadIdx.x;
    float acc = 0.f;
    for (int k = 0; k < plen; k++) {
        int l = t * pstr + k;
        float xv = xn[(size_t)n * LSEQ + (l < LSEQ ? l : LSEQ - 1)];  // right repl. pad
        acc += xv * valw[k * DMODEL + d];
    }
    float div = __expf((float)(d & ~1) * (-0.07195578575f)); // -ln(1e4)/128
    float arg = (float)t * div;
    float pe  = (d & 1) ? __cosf(arg) : __sinf(arg);
    z[((size_t)n * Pn + t) * DMODEL + d] = acc + pe;
}

// =====================================================================
// LayerNorm over D=128 -> bf16
// =====================================================================
__global__ __launch_bounds__(128) void ln_kernel(
    const float* __restrict__ z, unsigned short* __restrict__ zl,
    const float* __restrict__ g, const float* __restrict__ bta)
{
    const int row = blockIdx.x, d = threadIdx.x;
    __shared__ float sh[128];
    float v = z[(size_t)row * DMODEL + d];
    sh[d] = v; __syncthreads();
    for (int o = 64; o; o >>= 1) { if (d < o) sh[d] += sh[d + o]; __syncthreads(); }
    float mean = sh[0] / (float)DMODEL; __syncthreads();
    float dv = v - mean;
    sh[d] = dv * dv; __syncthreads();
    for (int o = 64; o; o >>= 1) { if (d < o) sh[d] += sh[d + o]; __syncthreads(); }
    float inv = rsqrtf(sh[0] / (float)DMODEL + 1e-5f);
    zl[(size_t)row * DMODEL + d] = f2bfu(dv * inv * g[d] + bta[d]);
}

// =====================================================================
// Causal depthwise conv (K=4) + SiLU on xz[:, :256] -> xc (bf16)
// =====================================================================
__global__ __launch_bounds__(256) void conv_silu_kernel(
    const unsigned short* __restrict__ xz, const float* __restrict__ cw,
    const float* __restrict__ cb, unsigned short* __restrict__ xc, int Ln)
{
    const int n = blockIdx.x, d = threadIdx.x;
    const float w0 = cw[d * DCONV + 0], w1 = cw[d * DCONV + 1];
    const float w2 = cw[d * DCONV + 2], w3 = cw[d * DCONV + 3];
    const float bb = cb[d];
    float xm3 = 0.f, xm2 = 0.f, xm1 = 0.f;
    for (int t = 0; t < Ln; t++) {
        float x0  = bf2f(xz[((size_t)n * Ln + t) * 512 + d]);
        float acc = w0 * xm3 + w1 * xm2 + w2 * xm1 + w3 * x0 + bb;
        float sil = acc / (1.f + __expf(-acc));
        xc[((size_t)n * Ln + t) * DINNER + d] = f2bfu(sil);
        xm3 = xm2; xm2 = xm1; xm1 = x0;
    }
}

// =====================================================================
// Selective scan, fused: one (seq, channel) lane per thread, 16-state in
// VGPRs; dt = softplus(dbl[:8] @ dt_w + dt_b) in-flight; D-residual and
// SiLU(zg) gate fused. dbl rows padded to stride 64.
// =====================================================================
__global__ __launch_bounds__(256) void scan_kernel(
    const float* __restrict__ dbl, const unsigned short* __restrict__ xc,
    const unsigned short* __restrict__ xz, const float* __restrict__ dtw,
    const float* __restrict__ dtb, const float* __restrict__ Alog,
    const float* __restrict__ Dp, unsigned short* __restrict__ yb, int Ln)
{
    const int n = blockIdx.x, d = threadIdx.x;
    __shared__ float row[40];
    float wdt[DTRANK];
#pragma unroll
    for (int r = 0; r < DTRANK; r++) wdt[r] = dtw[r * DINNER + d];
    float Aa[DSTATE];
#pragma unroll
    for (int s = 0; s < DSTATE; s++) Aa[s] = -__expf(Alog[d * DSTATE + s]);
    const float db = dtb[d], Dv = Dp[d];
    float h[DSTATE];
#pragma unroll
    for (int s = 0; s < DSTATE; s++) h[s] = 0.f;

    for (int t = 0; t < Ln; t++) {
        if (d < 40) row[d] = dbl[((size_t)n * Ln + t) * XPNP + d];
        __syncthreads();
        float dt = db;
#pragma unroll
        for (int r = 0; r < DTRANK; r++) dt += row[r] * wdt[r];
        dt = (dt > 20.f) ? dt : log1pf(__expf(dt));   // softplus
        float xt = bf2f(xc[((size_t)n * Ln + t) * DINNER + d]);
        float dx = dt * xt;
        float y  = 0.f;
#pragma unroll
        for (int s = 0; s < DSTATE; s++) {
            h[s] = h[s] * __expf(dt * Aa[s]) + dx * row[8 + s];
            y   += h[s] * row[24 + s];
        }
        y += xt * Dv;
        float zg = bf2f(xz[((size_t)n * Ln + t) * 512 + DINNER + d]);
        y *= zg / (1.f + __expf(-zg));                 // * silu(zg)
        yb[((size_t)n * Ln + t) * DINNER + d] = f2bfu(y);
        __syncthreads();
    }
}

// =====================================================================
// SE gate + concat into bf16 head-input buffer enc[n][d*224 + toff + t]
// =====================================================================
__global__ __launch_bounds__(128) void se_gate_kernel(
    const float* __restrict__ z, const float* __restrict__ w1,
    const float* __restrict__ b1, const float* __restrict__ w2,
    const float* __restrict__ b2, unsigned short* __restrict__ encbf,
    int Pn, int toff)
{
    const int n = blockIdx.x, d = threadIdx.x;
    __shared__ float s[4 * DMODEL];
    __shared__ float hh[SEH];
    const float eps = 1e-6f;

    float z0 = z[((size_t)n * Pn + 0) * DMODEL + d];
    float sum_z = z0, prev = z0;
    float sum_r2 = 0.f, sum_ar = 0.f, max_ar = 0.f, sum_bv = 0.f, prev_ar = 0.f;
    for (int t = 1; t < Pn; t++) {
        float zt = z[((size_t)n * Pn + t) * DMODEL + d];
        sum_z += zt;
        float r = zt - prev, ar = fabsf(r);
        sum_r2 += r * r; sum_ar += ar;
        max_ar = fmaxf(max_ar, ar);
        if (t >= 2) sum_bv += ar * prev_ar;
        prev = zt; prev_ar = ar;
    }
    float mean_z = sum_z / (float)Pn;
    float rv   = sqrtf(sum_r2 / (float)(Pn - 1) + eps);
    float bv   = sum_bv / (float)(Pn - 2);
    float jump = fmaxf(rv * rv - bv, 0.f);
    float tail = max_ar / (sum_ar / (float)(Pn - 1) + eps);
    s[d] = mean_z; s[DMODEL + d] = rv; s[2 * DMODEL + d] = jump; s[3 * DMODEL + d] = tail;
    __syncthreads();

    if (d < SEH) {
        float a = b1[d];
        for (int k = 0; k < 4 * DMODEL; k++) a += s[k] * w1[k * SEH + d];
        hh[d] = 0.5f * a * (1.f + erff(a * 0.70710678f));   // exact gelu
    }
    __syncthreads();

    float gd = b2[d];
#pragma unroll
    for (int j = 0; j < SEH; j++) gd += hh[j] * w2[j * DMODEL + d];
    float gate = 1.f / (1.f + __expf(-gd));
    for (int t = 0; t < Pn; t++) {
        float zt = z[((size_t)n * Pn + t) * DMODEL + d];
        encbf[(size_t)n * HEADNF + d * SUMP + toff + t] = f2bfu(zt * gate);
    }
}

// =====================================================================
// final: out[b][t][v] = (dec[n][t] + head_b[t]) * std[n] + mean[n]
// dec has padded row stride PREDP.
// =====================================================================
__global__ __launch_bounds__(PRED) void final_kernel(
    const float* __restrict__ dec, const float* __restrict__ hb,
    const float* __restrict__ meanv, const float* __restrict__ stdv,
    float* __restrict__ out)
{
    const int n = blockIdx.x, t = threadIdx.x;
    const int b = n / VVARS, v = n % VVARS;
    float val = (dec[(size_t)n * PREDP + t] + hb[t]) * stdv[n] + meanv[n];
    out[((size_t)b * PRED + t) * VVARS + v] = val;
}

// =====================================================================
// host driver
// =====================================================================
extern "C" void kernel_launch(void* const* d_in, const int* in_sizes, int n_in,
                              void* d_out, int out_size, void* d_ws, size_t ws_size,
                              hipStream_t stream)
{
    // JAX pytree (sorted-key) flattening of {'params':..., 'x_enc':...}:
    // 0..10  enc[0]: A_log,D,conv_b,conv_w,dt_b,dt_w,in_w,ln_b,ln_g,out_w,xproj_w
    // 11..21 enc[1] (same order)
    // 22..33 gate[0..2]: b1,b2,w1,w2
    // 34 head_b  35 head_w  36..38 val_w[0..2]  39 x_enc
    auto F = [&](int i) -> const float* { return (const float*)d_in[i]; };
    const float* x_enc  = F(39);
    const float* head_b = F(34);
    const float* head_w = F(35);

    const int PLEN[3] = {8, 16, 32};
    const int PSTR[3] = {4, 8, 16};
    const int PNUM[3] = {128, 64, 32};
    const int POFF[3] = {0, 128, 192};

    // workspace carve-out (256B aligned blocks)
    char* base = (char*)d_ws;
    size_t off = 0;
    auto alloc = [&](size_t bytes) -> void* {
        off = (off + 255) & ~(size_t)255;
        void* p = base + off;
        off += bytes;
        return p;
    };
    float* meanv = (float*)alloc(NSEQ * 4);
    float* stdv  = (float*)alloc(NSEQ * 4);
    float* xn    = (float*)alloc((size_t)NSEQ * LSEQ * 4);
    const size_t MAXTOK = (size_t)NSEQ * 128;  // biggest scale (Pn=128)
    float*          z    = (float*)alloc(MAXTOK * DMODEL * 4);
    unsigned short* zl   = (unsigned short*)alloc(MAXTOK * DMODEL * 2);
    unsigned short* xz   = (unsigned short*)alloc(MAXTOK * 512 * 2);
    unsigned short* xc   = (unsigned short*)alloc(MAXTOK * DINNER * 2);
    float*          dbl  = (float*)alloc(MAXTOK * XPNP * 4);
    unsigned short* ybuf = (unsigned short*)alloc(MAXTOK * DINNER * 2);
    unsigned short* encbf = (unsigned short*)alloc((size_t)NSEQ * HEADNF * 2);
    float*          dec  = (float*)alloc((size_t)NSEQ * PREDP * 4);
    unsigned short* inw_bf[2]  = {(unsigned short*)alloc(DMODEL * 512 * 2),
                                  (unsigned short*)alloc(DMODEL * 512 * 2)};
    unsigned short* xpw_bf[2]  = {(unsigned short*)alloc(DINNER * XPNP * 2),
                                  (unsigned short*)alloc(DINNER * XPNP * 2)};
    unsigned short* outw_bf[2] = {(unsigned short*)alloc(DINNER * DMODEL * 2),
                                  (unsigned short*)alloc(DINNER * DMODEL * 2)};
    unsigned short* headw_bf   = (unsigned short*)alloc((size_t)HEADNF * PREDP * 2);

    // ---- instance stats + normalized transposed series
    normalize_kernel<<<NSEQ, 256, 0, stream>>>(x_enc, xn, meanv, stdv);

    // ---- weight conversion to bf16 (zero-padded where N was padded)
    for (int l = 0; l < 2; l++) {
        int bb = 11 * l;
        convert_pad_bf16<<<(DMODEL * 512 + 255) / 256, 256, 0, stream>>>(
            F(bb + 6), inw_bf[l], DMODEL, 512, 512);
        convert_pad_bf16<<<(DINNER * XPNP + 255) / 256, 256, 0, stream>>>(
            F(bb + 10), xpw_bf[l], DINNER, XPN, XPNP);
        convert_pad_bf16<<<(DINNER * DMODEL + 255) / 256, 256, 0, stream>>>(
            F(bb + 9), outw_bf[l], DINNER, DMODEL, DMODEL);
    }
    convert_pad_bf16<<<((size_t)HEADNF * PREDP + 255) / 256, 256, 0, stream>>>(
        head_w, headw_bf, HEADNF, PRED, PREDP);

    // ---- three patch scales
    for (int i = 0; i < 3; i++) {
        const int Pn  = PNUM[i];
        const int tok = NSEQ * Pn;          // multiple of 64

        patch_embed_kernel<<<tok, DMODEL, 0, stream>>>(xn, F(36 + i), z, Pn, PLEN[i], PSTR[i]);

        for (int l = 0; l < 2; l++) {
            int bb = 11 * l;
            const float* A_log  = F(bb + 0);
            const float* Dparam = F(bb + 1);
            const float* conv_b = F(bb + 2);
            const float* conv_w = F(bb + 3);
            const float* dt_b   = F(bb + 4);
            const float* dt_w   = F(bb + 5);
            const float* ln_b   = F(bb + 7);
            const float* ln_g   = F(bb + 8);

            ln_kernel<<<tok, DMODEL, 0, stream>>>(z, zl, ln_g, ln_b);

            // xz = LN(z) @ in_w : [tok,128] x [128,512] -> bf16 [tok,512]
            gemm_bf16_kernel<<<dim3(512 / 64, tok / 64), 256, 0, stream>>>(
                zl, DMODEL, inw_bf[l], 512, nullptr, xz, 512, DMODEL, 2);

            conv_silu_kernel<<<NSEQ, DINNER, 0, stream>>>(xz, conv_w, conv_b, xc, Pn);

            // dbl = xc @ xproj : [tok,256] x [256,64pad] -> f32 [tok,64]
            gemm_bf16_kernel<<<dim3(XPNP / 64, tok / 64), 256, 0, stream>>>(
                xc, DINNER, xpw_bf[l], XPNP, dbl, nullptr, XPNP, DINNER, 0);

            scan_kernel<<<NSEQ, DINNER, 0, stream>>>(dbl, xc, xz, dt_w, dt_b,
                                                     A_log, Dparam, ybuf, Pn);

            // z += y @ out_w : [tok,256] x [256,128] accumulate residual
            gemm_bf16_kernel<<<dim3(DMODEL / 64, tok / 64), 256, 0, stream>>>(
                ybuf, DINNER, outw_bf[l], DMODEL, z, nullptr, DMODEL, DINNER, 1);
        }

        int gb = 22 + 4 * i;  // b1,b2,w1,w2
        se_gate_kernel<<<NSEQ, DMODEL, 0, stream>>>(z, F(gb + 2), F(gb + 0),
                                                    F(gb + 3), F(gb + 1),
                                                    encbf, Pn, POFF[i]);
    }

    // ---- head: dec = enc_flat @ head_w : [512,28672] x [28672,128pad]
    gemm_bf16_kernel<<<dim3(PREDP / 64, NSEQ / 64), 256, 0, stream>>>(
        encbf, HEADNF, headw_bf, PREDP, dec, nullptr, PREDP, HEADNF, 0);

    // ---- denormalize + transpose to [B, PRED, V]
    final_kernel<<<NSEQ, PRED, 0, stream>>>(dec, head_b, meanv, stdv, (float*)d_out);
}
